// Encoder_40286793236574
// MI455X (gfx1250) — matrix operand order repaired and checked by reading, not compiled, not run on hardware
//
#include <hip/hip_runtime.h>

// Problem dims (fixed by the reference)
#define V_SZ   32000
#define B_SZ   64
#define T_LEN  512
#define E_DIM  512
#define H_DIM  512
#define FOURH  2048
#define KT     32      // 32 k-tiles of 32 k each

typedef __attribute__((ext_vector_type(16))) __bf16    v16bf;
typedef __attribute__((ext_vector_type(8)))  float     v8f;
typedef __attribute__((ext_vector_type(8)))  unsigned  v8u;

static __device__ __forceinline__ unsigned f32_to_bf16_rne(float f) {
  unsigned u = __builtin_bit_cast(unsigned, f);
  return (u + 0x7FFFu + ((u >> 16) & 1u)) >> 16;
}
static __device__ __forceinline__ unsigned pack2_bf16(float lo, float hi) {
  return f32_to_bf16_rne(lo) | (f32_to_bf16_rne(hi) << 16);
}
static __device__ __forceinline__ float sigmoid_fast(float x) {
  return 1.0f / (1.0f + __expf(-x));
}
static __device__ __forceinline__ float tanh_fast(float x) {
  return 1.0f - 2.0f / (1.0f + __expf(2.0f * x));
}
static __device__ __forceinline__ v8f wmma_bf16(v16bf a, v16bf b, v8f c) {
  // D = A(16x32 bf16) * B(32x16 bf16) + C(16x16 f32)
  return __builtin_amdgcn_wmma_f32_16x16x32_bf16(
      /*neg_a=*/false, a, /*neg_b=*/false, b,
      /*c_mod=*/(short)0, c, /*reuse_a=*/false, /*reuse_b=*/false);
}

// B tile: packed layout [nt][kt][lane][8 u32]; two 16B loads per lane.
static __device__ __forceinline__ v16bf load_b_tile(const unsigned* __restrict__ P,
                                                    int nt, int kt, int lane) {
  const uint4* p = (const uint4*)(P + ((((nt * KT) + kt) * 32 + lane) << 3));
  uint4 x = p[0];
  uint4 y = p[1];
  v8u u;
  u[0] = x.x; u[1] = x.y; u[2] = x.z; u[3] = x.w;
  u[4] = y.x; u[5] = y.y; u[6] = y.z; u[7] = y.w;
  return __builtin_bit_cast(v16bf, u);
}

// A tile chunk from bf16 memory: 8 bf16 at p (K k..k+7) + 8 bf16 at p+16
// (K k+16..k+23), matching the 16-bit A VGPR layout (ISA 7.12.2).
static __device__ __forceinline__ v16bf a_from_bf16(const unsigned short* __restrict__ p) {
  uint4 x = ((const uint4*)p)[0];
  uint4 y = ((const uint4*)(p + 16))[0];
  v8u u;
  u[0] = x.x; u[1] = x.y; u[2] = x.z; u[3] = x.w;
  u[4] = y.x; u[5] = y.y; u[6] = y.z; u[7] = y.w;
  return __builtin_bit_cast(v16bf, u);
}

// Fallback: A tile chunk converted from fp32 on the fly.
static __device__ __forceinline__ v16bf a_from_f32(const float* __restrict__ p) {
  float4 a0 = ((const float4*)p)[0];
  float4 a1 = ((const float4*)p)[1];
  float4 b0 = ((const float4*)(p + 16))[0];
  float4 b1 = ((const float4*)(p + 16))[1];
  v8u u;
  u[0] = pack2_bf16(a0.x, a0.y); u[1] = pack2_bf16(a0.z, a0.w);
  u[2] = pack2_bf16(a1.x, a1.y); u[3] = pack2_bf16(a1.z, a1.w);
  u[4] = pack2_bf16(b0.x, b0.y); u[5] = pack2_bf16(b0.z, b0.w);
  u[6] = pack2_bf16(b1.x, b1.y); u[7] = pack2_bf16(b1.z, b1.w);
  return __builtin_bit_cast(v16bf, u);
}

// ---------------------------------------------------------------------------
// Pack a fp32 [512 x 2048] row-major weight matrix into bf16 WMMA B-tile
// layout: P[((nt*KT + kt)*32 + lane)*8 + v] = pair(W[k][n], W[k+1][n]) with
// n = nt*16 + (lane&15), k = kt*32 + (lane>=16 ? 8 : 0) + 2v + (v>=4 ? 8 : 0).  524288 thr.
// ---------------------------------------------------------------------------
__global__ __launch_bounds__(256)
void pack_w_kernel(const float* __restrict__ W, unsigned* __restrict__ P) {
  int idx  = blockIdx.x * 256 + threadIdx.x;     // < 524288
  int v    = idx & 7;
  int lane = (idx >> 3) & 31;
  int kt   = (idx >> 8) & 31;
  int nt   = idx >> 13;
  int n = nt * 16 + (lane & 15);
  int k = kt * 32 + ((lane >= 16) ? 8 : 0) + 2 * v + ((v >= 4) ? 8 : 0);
  float f0 = W[k * FOURH + n];
  float f1 = W[(k + 1) * FOURH + n];
  P[idx] = pack2_bf16(f0, f1);
}

// Convert the whole embedding table fp32 -> bf16 (row-major preserved).
__global__ __launch_bounds__(256)
void cvt_emb_kernel(const float* __restrict__ E, unsigned* __restrict__ Ebf) {
  int i = blockIdx.x * 256 + threadIdx.x;        // < 8192000
  float2 f = ((const float2*)E)[i];
  Ebf[i] = pack2_bf16(f.x, f.y);
}

__global__ __launch_bounds__(256)
void init_state_kernel(float* __restrict__ c_st, float* __restrict__ h_st,
                       unsigned short* __restrict__ h_bf) {
  int i = blockIdx.x * 256 + threadIdx.x;        // < 32768
  c_st[i] = 0.0f;
  h_st[i] = 0.0f;
  h_bf[i] = 0;
}

// ---------------------------------------------------------------------------
// One LSTM timestep. 16 blocks x 256 threads = 128 waves; wave w owns the
// 16x16 (batch x hidden) patch (b0 = (w/32)*16, h0 = (w%32)*16) and all four
// gate columns {h0, h0+512, h0+1024, h0+1536} of z = emb[x_t]@Wi + h@Wh + b.
// h_bf is double-buffered across timesteps (read t-parity, write other).
// Loop bodies are uniform (unconditional speculative prefetch) and all four
// gate B-tiles are kept live per k-step so the scheduler can stagger
// s_wait_loadcnt and keep accumulators pinned in place.
// ---------------------------------------------------------------------------
template <bool EMB_BF16>
__global__ __launch_bounds__(256)
void lstm_step_kernel(const int* __restrict__ inputs,
                      const int* __restrict__ lengths,
                      const float* __restrict__ emb_f32,
                      const unsigned short* __restrict__ emb_bf,
                      const float* __restrict__ bias,
                      const unsigned* __restrict__ WiP,
                      const unsigned* __restrict__ WhP,
                      float* __restrict__ c_st, float* __restrict__ h_st,
                      const unsigned short* __restrict__ h_bf_r,
                      unsigned short* __restrict__ h_bf_w,
                      float* __restrict__ out, int t) {
  const int lane = threadIdx.x & 31;
  const int wave = blockIdx.x * 8 + (threadIdx.x >> 5);   // 0..127
  const int b0   = (wave >> 5) << 4;                      // batch tile base
  const int h0   = (wave & 31) << 4;                      // hidden tile base
  const int ln   = lane & 15;
  const bool hi  = lane >= 16;
  const int arow = b0 + ln;                               // A-matrix row (batch)
  const int ntb  = h0 >> 4;                               // base n-tile in a gate

  // Gate accumulators, bias-initialized (same value down the 8 row-VGPRs).
  v8f acc0, acc1, acc2, acc3;
  {
    float bv0 = bias[0 * H_DIM + h0 + ln];
    float bv1 = bias[1 * H_DIM + h0 + ln];
    float bv2 = bias[2 * H_DIM + h0 + ln];
    float bv3 = bias[3 * H_DIM + h0 + ln];
#pragma unroll
    for (int r = 0; r < 8; ++r) {
      acc0[r] = bv0; acc1[r] = bv1; acc2[r] = bv2; acc3[r] = bv3;
    }
  }

  // ---- Phase 1: x_t @ Wi ----
  const int tok = inputs[arow * T_LEN + t];
  {
    const unsigned short* erow_bf = emb_bf + tok * E_DIM + (hi ? 8 : 0);
    const float*          erow_f  = emb_f32 + tok * E_DIM + (hi ? 8 : 0);
    for (int kt = 0; kt < KT; ++kt) {
      v16bf A = EMB_BF16 ? a_from_bf16(erow_bf + kt * 32)
                         : a_from_f32(erow_f + kt * 32);
      v16bf B0 = load_b_tile(WiP, 0 * 32 + ntb, kt, lane);
      v16bf B1 = load_b_tile(WiP, 1 * 32 + ntb, kt, lane);
      v16bf B2 = load_b_tile(WiP, 2 * 32 + ntb, kt, lane);
      v16bf B3 = load_b_tile(WiP, 3 * 32 + ntb, kt, lane);
      // Speculative prefetch of next k-tile (harmless past the end).
      __builtin_prefetch(WiP + ((((ntb * KT) + kt + 1) * 32 + lane) << 3), 0, 1);
      acc0 = wmma_bf16(A, B0, acc0);
      acc1 = wmma_bf16(A, B1, acc1);
      acc2 = wmma_bf16(A, B2, acc2);
      acc3 = wmma_bf16(A, B3, acc3);
    }
  }

  // ---- Phase 2: h_{t-1} @ Wh ----
  {
    const unsigned short* hrow = h_bf_r + arow * H_DIM + (hi ? 8 : 0);
    for (int kt = 0; kt < KT; ++kt) {
      v16bf A  = a_from_bf16(hrow + kt * 32);
      v16bf B0 = load_b_tile(WhP, 0 * 32 + ntb, kt, lane);
      v16bf B1 = load_b_tile(WhP, 1 * 32 + ntb, kt, lane);
      v16bf B2 = load_b_tile(WhP, 2 * 32 + ntb, kt, lane);
      v16bf B3 = load_b_tile(WhP, 3 * 32 + ntb, kt, lane);
      __builtin_prefetch(WhP + ((((ntb * KT) + kt + 1) * 32 + lane) << 3), 0, 1);
      acc0 = wmma_bf16(A, B0, acc0);
      acc1 = wmma_bf16(A, B1, acc1);
      acc2 = wmma_bf16(A, B2, acc2);
      acc3 = wmma_bf16(A, B3, acc3);
    }
  }

  // ---- Gates + length masking. C/D layout: VGPR r, lanes<16 -> M=r,
  // lanes>=16 -> M=r+8; N = lane&15. Each lane owns 8 state elements. ----
#pragma unroll
  for (int r = 0; r < 8; ++r) {
    const int bidx = b0 + r + (hi ? 8 : 0);
    const int hcol = h0 + ln;
    const int sidx = bidx * H_DIM + hcol;
    float iv = acc0[r], fv = acc1[r], gv = acc2[r], ov = acc3[r];
    float ig = sigmoid_fast(iv);
    float fg = sigmoid_fast(fv);
    float og = sigmoid_fast(ov);
    float gt = tanh_fast(gv);
    float cold = c_st[sidx];
    float nc   = fg * cold + ig * gt;
    float nh   = og * tanh_fast(nc);
    bool  m    = t < lengths[bidx];
    float cn = m ? nc : cold;
    float hn = m ? nh : h_st[sidx];
    c_st[sidx]   = cn;
    h_st[sidx]   = hn;
    h_bf_w[sidx] = (unsigned short)f32_to_bf16_rne(hn);
    out[bidx * (T_LEN * H_DIM) + t * H_DIM + hcol] = hn;   // hs[b][t][h]
  }
}

__global__ __launch_bounds__(256)
void final_copy_kernel(const float* __restrict__ c_st,
                       const float* __restrict__ h_st,
                       float* __restrict__ out) {
  int i = blockIdx.x * 256 + threadIdx.x;                  // < 32768
  const int base = B_SZ * T_LEN * H_DIM;                   // 16777216
  out[base + i]                = c_st[i];                  // c_f
  out[base + B_SZ * H_DIM + i] = h_st[i];                  // h_f
}

extern "C" void kernel_launch(void* const* d_in, const int* in_sizes, int n_in,
                              void* d_out, int out_size, void* d_ws, size_t ws_size,
                              hipStream_t stream) {
  (void)in_sizes; (void)n_in; (void)out_size;
  const int*   inputs  = (const int*)d_in[0];    // [B,T] int32
  const int*   lengths = (const int*)d_in[1];    // [B]   int32
  const float* emb     = (const float*)d_in[2];  // [V,E] f32
  const float* Wi      = (const float*)d_in[3];  // [E,4H] f32
  const float* Wh      = (const float*)d_in[4];  // [H,4H] f32
  const float* bias    = (const float*)d_in[5];  // [4H]  f32
  float* out = (float*)d_out;

  // Workspace layout
  const size_t OFF_WIP = 0;                              // 2 MB (524288 u32)
  const size_t OFF_WHP = 2u << 20;                       // 2 MB
  const size_t OFF_C   = 4u << 20;                       // 128 KB
  const size_t OFF_H   = (4u << 20) + (128u << 10);      // 128 KB
  const size_t OFF_HBA = (4u << 20) + (256u << 10);      // 64 KB bf16
  const size_t OFF_HBB = (4u << 20) + (320u << 10);      // 64 KB bf16
  const size_t OFF_EMB = (4u << 20) + (384u << 10);      // 31.25 MB bf16 table
  const size_t NEED_SMALL = OFF_EMB;
  const size_t NEED_BIG   = OFF_EMB + (size_t)V_SZ * E_DIM * 2;
  if (ws_size < NEED_SMALL) return;
  const bool use_bf_emb = (ws_size >= NEED_BIG);

  char* ws = (char*)d_ws;
  unsigned*       WiP  = (unsigned*)(ws + OFF_WIP);
  unsigned*       WhP  = (unsigned*)(ws + OFF_WHP);
  float*          c_st = (float*)(ws + OFF_C);
  float*          h_st = (float*)(ws + OFF_H);
  unsigned short* hbA  = (unsigned short*)(ws + OFF_HBA);
  unsigned short* hbB  = (unsigned short*)(ws + OFF_HBB);
  unsigned short* ebf  = (unsigned short*)(ws + OFF_EMB);

  pack_w_kernel<<<2048, 256, 0, stream>>>(Wi, WiP);
  pack_w_kernel<<<2048, 256, 0, stream>>>(Wh, WhP);
  init_state_kernel<<<128, 256, 0, stream>>>(c_st, h_st, hbA);
  if (use_bf_emb) {
    // 32000*512/2 u32 elements / 256 threads = 32000 blocks
    cvt_emb_kernel<<<32000, 256, 0, stream>>>(emb, (unsigned*)ebf);
  }

  for (int t = 0; t < T_LEN; ++t) {
    const unsigned short* hr = (t & 1) ? hbB : hbA;
    unsigned short*       hw = (t & 1) ? hbA : hbB;
    if (use_bf_emb) {
      lstm_step_kernel<true><<<16, 256, 0, stream>>>(
          inputs, lengths, emb, ebf, bias, WiP, WhP, c_st, h_st, hr, hw, out, t);
    } else {
      lstm_step_kernel<false><<<16, 256, 0, stream>>>(
          inputs, lengths, emb, ebf, bias, WiP, WhP, c_st, h_st, hr, hw, out, t);
    }
  }

  final_copy_kernel<<<128, 256, 0, stream>>>(c_st, h_st, out);
}
